// CausalSelfAttention_14242111553573
// MI455X (gfx1250) — matrix operand order repaired and checked by reading, not compiled, not run on hardware
//
#include <hip/hip_runtime.h>
#include <hip/hip_bf16.h>

// ---------------- problem constants (match reference) ----------------
#define BB   2
#define TT   2048
#define EE   1024
#define HH   16
#define HKV  8
#define DD   64
#define WIN  1024
#define NEGF (-1e30f)
#define RMS_EPS 1.1920929e-7f

typedef __bf16 bf16;
typedef __attribute__((ext_vector_type(16))) __bf16 bf16x16;
typedef __attribute__((ext_vector_type(8)))  __bf16 bf16x8;
typedef __attribute__((ext_vector_type(8)))  float  f32x8;

// ---------------- WMMA helpers (CDNA5: v_wmma_f32_16x16x32_bf16) ------
__device__ __forceinline__ f32x8 wmma_bf16(bf16x16 a, bf16x16 b, f32x8 c) {
  // (neg_a, A, neg_b, B, c_mod, C, reuse_a, reuse_b)
  return __builtin_amdgcn_wmma_f32_16x16x32_bf16(false, a, false, b, (short)0, c,
                                                 false, false);
}

// A-fragment (16x32, row-major source, leading dim ld elements).
// ISA layout: lanes 0-15 hold M=lane; halves 0..7 -> K = k0 + 8*(lane>=16)
// .. contiguous; halves 8..15 -> K = k0+16+8*(lane>=16) .. contiguous.
__device__ __forceinline__ bf16x16 load_a_frag(const bf16* base, int ld, int lane) {
  int m  = lane & 15;
  int ko = (lane >> 4) * 8;
  const bf16* r = base + (size_t)m * ld;
  bf16x8 lo = *(const bf16x8*)(r + ko);
  bf16x8 hi = *(const bf16x8*)(r + 16 + ko);
  bf16x16 a;
#pragma unroll
  for (int i = 0; i < 8; ++i) { a[i] = lo[i]; a[8 + i] = hi[i]; }
  return a;
}

// B-fragment for D = A x B where B = sourceT. Source is row-major [16 x 32]
// (N rows x K cols, leading dim ld). ISA layout: lane holds column N=lane&15,
// halves 0..7 -> K = 8*(lane>=16) + h; halves 8..15 -> K = 16 + 8*(lane>=16) + (h-8).
__device__ __forceinline__ bf16x16 load_b_frag(const bf16* base, int ld, int lane) {
  int n  = lane & 15;
  int ko = (lane >> 4) * 8;
  const bf16* r = base + (size_t)n * ld + ko;
  bf16x8 lo = *(const bf16x8*)(r);
  bf16x8 hi = *(const bf16x8*)(r + 16);
  bf16x16 b;
#pragma unroll
  for (int i = 0; i < 8; ++i) { b[i] = lo[i]; b[8 + i] = hi[i]; }
  return b;
}

// ---------------- fp32 -> bf16 convert ----------------
__global__ void cvt_bf16_kernel(const float* __restrict__ in,
                                bf16* __restrict__ out, int n) {
  int i = blockIdx.x * blockDim.x + threadIdx.x;
  int stride = gridDim.x * blockDim.x;
  for (; i < n; i += stride) out[i] = (bf16)in[i];
}

// ---------------- generic GEMM: C(MxN) = A(MxK) * W(NxK)^T --------------
// bf16 inputs, fp32 out. Block = 256 threads = 8 waves; block tile 128x128;
// wave tile 32x64 (2x4 wmma C-frags). Fragments fed from global (L2 = 192MB
// holds the whole working set on MI455X).
__global__ __launch_bounds__(256) void gemm_bf16_nt(
    const bf16* __restrict__ A, const bf16* __restrict__ W,
    float* __restrict__ C, int M, int N, int K) {
  int lane = threadIdx.x & 31;
  int wave = threadIdx.x >> 5;
  int mw = wave & 3, nw = wave >> 2;
  int m0 = blockIdx.y * 128 + mw * 32;
  int n0 = blockIdx.x * 128 + nw * 64;

  f32x8 c[2][4] = {};

  for (int k0 = 0; k0 < K; k0 += 32) {
    bf16x16 a0 = load_a_frag(A + (size_t)m0 * K + k0, K, lane);
    bf16x16 a1 = load_a_frag(A + (size_t)(m0 + 16) * K + k0, K, lane);
#pragma unroll
    for (int j = 0; j < 4; ++j) {
      bf16x16 b = load_b_frag(W + (size_t)(n0 + 16 * j) * K + k0, K, lane);
      c[0][j] = wmma_bf16(a0, b, c[0][j]);
      c[1][j] = wmma_bf16(a1, b, c[1][j]);
    }
  }
  // C-frag layout: vgpr r -> M = r + 8*(lane>=16), N = lane&15
  int rlo = lane & 15, rhi = (lane >> 4) * 8;
#pragma unroll
  for (int i = 0; i < 2; ++i)
#pragma unroll
    for (int j = 0; j < 4; ++j)
#pragma unroll
      for (int r = 0; r < 8; ++r)
        C[(size_t)(m0 + i * 16 + r + rhi) * N + n0 + j * 16 + rlo] = c[i][j][r];
}

// ---------------- RoPE + RMSNorm on q (per b,t,h row of 64) -------------
__global__ void qpost_kernel(const float* __restrict__ qraw,
                             const float* __restrict__ cosp,
                             const float* __restrict__ sinp,
                             bf16* __restrict__ qout) {
  int gid = blockIdx.x * blockDim.x + threadIdx.x;  // B*T*H
  int h = gid % HH;
  int bt = gid / HH;
  int t = bt % TT;
  const float* src = qraw + (size_t)bt * (HH * DD) + h * DD;
  const float* cs = cosp + (size_t)t * (DD / 2);
  const float* sn = sinp + (size_t)t * (DD / 2);
  float o1[32], o2[32];
  float ss = 0.f;
#pragma unroll
  for (int d = 0; d < 32; ++d) {
    float x1 = src[d], x2 = src[d + 32];
    float c = cs[d], s = sn[d];
    o1[d] = x1 * c + x2 * s;
    o2[d] = -x1 * s + x2 * c;
    ss += o1[d] * o1[d] + o2[d] * o2[d];
  }
  float r = rsqrtf(ss * (1.0f / 64.0f) + RMS_EPS);
  bf16* dst = qout + (size_t)bt * (HH * DD) + h * DD;
#pragma unroll
  for (int d = 0; d < 32; ++d) {
    dst[d]      = (bf16)(o1[d] * r);
    dst[d + 32] = (bf16)(o2[d] * r);
  }
}

// ---------- K: RoPE+RMSNorm; V: gate(ve) add; per (b,t,kv-head) --------
__global__ void kvpost_kernel(const float* __restrict__ kraw,
                              const float* __restrict__ vraw,
                              const float* __restrict__ x,
                              const float* __restrict__ ve,
                              const float* __restrict__ Wgate,
                              const float* __restrict__ cosp,
                              const float* __restrict__ sinp,
                              bf16* __restrict__ kout,
                              bf16* __restrict__ vout) {
  int gid = blockIdx.x * blockDim.x + threadIdx.x;  // B*T*HKV
  int kh = gid % HKV;
  int bt = gid / HKV;
  int t = bt % TT;

  // --- K: rope + rmsnorm ---
  {
    const float* src = kraw + (size_t)bt * (HKV * DD) + kh * DD;
    const float* cs = cosp + (size_t)t * (DD / 2);
    const float* sn = sinp + (size_t)t * (DD / 2);
    float o1[32], o2[32];
    float ss = 0.f;
#pragma unroll
    for (int d = 0; d < 32; ++d) {
      float x1 = src[d], x2 = src[d + 32];
      float c = cs[d], s = sn[d];
      o1[d] = x1 * c + x2 * s;
      o2[d] = -x1 * s + x2 * c;
      ss += o1[d] * o1[d] + o2[d] * o2[d];
    }
    float r = rsqrtf(ss * (1.0f / 64.0f) + RMS_EPS);
    bf16* dst = kout + (size_t)bt * (HKV * DD) + kh * DD;
#pragma unroll
    for (int d = 0; d < 32; ++d) {
      dst[d]      = (bf16)(o1[d] * r);
      dst[d + 32] = (bf16)(o2[d] * r);
    }
  }
  // --- V: v = v_raw + 2*sigmoid(x[:32] . Wgate[kh]) * ve ---
  {
    const float* xr = x + (size_t)bt * EE;
    const float* wg = Wgate + kh * 32;
    float z = 0.f;
#pragma unroll
    for (int i = 0; i < 32; ++i) z += xr[i] * wg[i];
    float gate = 2.0f / (1.0f + __expf(-z));
    const float* vs = vraw + (size_t)bt * (HKV * DD) + kh * DD;
    const float* vg = ve + (size_t)bt * (HKV * DD) + kh * DD;
    bf16* dst = vout + (size_t)bt * (HKV * DD) + kh * DD;
#pragma unroll
    for (int d = 0; d < DD; ++d) dst[d] = (bf16)(vs[d] + gate * vg[d]);
  }
}

// ---------------- flash-style sliding-window attention -----------------
// grid = B*H*(T/128); block = 256 (8 waves). Wave owns a 16-query tile.
// Per 32-key iteration: K-tile + V^T-tile staged in LDS; S = Q K^T via 4
// wmma; online softmax (16-lane shuffle reductions); P -> LDS -> A-frag;
// O += P V via 4 wmma.
__global__ __launch_bounds__(256) void attn_kernel(
    const bf16* __restrict__ qb, const bf16* __restrict__ kb,
    const bf16* __restrict__ vb, bf16* __restrict__ yb) {
  __shared__ bf16 Ks[32][DD];       // key tile, row-major (key, d)
  __shared__ bf16 Vts[DD][32];      // V^T tile (d, key)
  __shared__ bf16 Ps[8][16][32];    // per-wave P staging

  int lane = threadIdx.x & 31;
  int wave = threadIdx.x >> 5;
  int blk = blockIdx.x;
  int qblk = blk % (TT / 128);
  int bh = blk / (TT / 128);
  int h = bh % HH;
  int b = bh / HH;
  int kvh = h >> 1;                 // GQA: H/HKV == 2
  int q0 = qblk * 128;
  int qt = q0 + wave * 16;

  // Q fragments for this wave's 16 rows (d 0..31, d 32..63)
  const bf16* qbase = qb + ((size_t)(b * TT + qt) * HH + h) * DD;
  bf16x16 qa0 = load_a_frag(qbase, HH * DD, lane);
  bf16x16 qa1 = load_a_frag(qbase + 32, HH * DD, lane);

  f32x8 o[4] = {};
  float mstat[8], lstat[8];
#pragma unroll
  for (int r = 0; r < 8; ++r) { mstat[r] = NEGF; lstat[r] = 0.f; }

  int ks0 = q0 - WIN + 1;
  if (ks0 < 0) ks0 = 0;
  ks0 &= ~31;
  int kend = q0 + 128;

  int hi = lane >> 4, lo = lane & 15;

  for (int kt = ks0; kt < kend; kt += 32) {
    __syncthreads();
    {  // cooperative staging: 256 threads move 32x64 K and V^T
      int row = threadIdx.x >> 3;        // 0..31 (key within tile)
      int dc = (threadIdx.x & 7) * 8;    // 0..56
      int key = kt + row;                // always in [0, T) by construction
      const bf16* kp = kb + ((size_t)(b * TT + key) * HKV + kvh) * DD + dc;
      *(bf16x8*)&Ks[row][dc] = *(const bf16x8*)kp;
      const bf16* vp = vb + ((size_t)(b * TT + key) * HKV + kvh) * DD + dc;
      bf16x8 vv = *(const bf16x8*)vp;
#pragma unroll
      for (int j = 0; j < 8; ++j) Vts[dc + j][row] = vv[j];
    }
    __syncthreads();

    // S tiles: s0 -> keys kt..kt+15, s1 -> keys kt+16..kt+31
    f32x8 s0 = {}, s1 = {};
    {
      bf16x16 bk;
      bk = load_b_frag(&Ks[0][0], DD, lane);        s0 = wmma_bf16(qa0, bk, s0);
      bk = load_b_frag(&Ks[0][0] + 32, DD, lane);   s0 = wmma_bf16(qa1, bk, s0);
      bk = load_b_frag(&Ks[16][0], DD, lane);       s1 = wmma_bf16(qa0, bk, s1);
      bk = load_b_frag(&Ks[16][0] + 32, DD, lane);  s1 = wmma_bf16(qa1, bk, s1);
    }

    int key0 = kt + lo, key1 = kt + 16 + lo;
#pragma unroll
    for (int r = 0; r < 8; ++r) {
      int qrow = qt + r + hi * 8;
      float v0 = s0[r] * 0.125f;  // D^-0.5
      float v1 = s1[r] * 0.125f;
      bool ok0 = (key0 <= qrow) && (key0 > qrow - WIN);
      bool ok1 = (key1 <= qrow) && (key1 > qrow - WIN);
      v0 = ok0 ? v0 : NEGF;
      v1 = ok1 ? v1 : NEGF;
      float tm = fmaxf(v0, v1);
#pragma unroll
      for (int off = 8; off >= 1; off >>= 1)
        tm = fmaxf(tm, __shfl_xor(tm, off, 32));
      float nm = fmaxf(mstat[r], tm);
      float alpha = __expf(mstat[r] - nm);
      float p0 = ok0 ? __expf(v0 - nm) : 0.f;
      float p1 = ok1 ? __expf(v1 - nm) : 0.f;
      float rs = p0 + p1;
#pragma unroll
      for (int off = 8; off >= 1; off >>= 1) rs += __shfl_xor(rs, off, 32);
      lstat[r] = lstat[r] * alpha + rs;
      mstat[r] = nm;
#pragma unroll
      for (int j = 0; j < 4; ++j) o[j][r] *= alpha;
      Ps[wave][r + hi * 8][lo] = (bf16)p0;
      Ps[wave][r + hi * 8][16 + lo] = (bf16)p1;
    }

    // O += P * V  (wave-private P; LDS ops are in order within a wave)
    bf16x16 pa = load_a_frag(&Ps[wave][0][0], 32, lane);
#pragma unroll
    for (int j = 0; j < 4; ++j) {
      bf16x16 bv = load_b_frag(&Vts[j * 16][0], 32, lane);
      o[j] = wmma_bf16(pa, bv, o[j]);
    }
  }

  // normalize rows by 1/l and emit bf16 y (B, T, H*D)
#pragma unroll
  for (int r = 0; r < 8; ++r) {
    float inv = 1.0f / lstat[r];  // l > 0: self key always valid
    int qrow = qt + r + hi * 8;
    bf16* yrow = yb + ((size_t)(b * TT + qrow) * HH + h) * DD;
#pragma unroll
    for (int j = 0; j < 4; ++j) yrow[j * 16 + lo] = (bf16)(o[j][r] * inv);
  }
}

// ------------------------------ launcher -------------------------------
extern "C" void kernel_launch(void* const* d_in, const int* in_sizes, int n_in,
                              void* d_out, int out_size, void* d_ws, size_t ws_size,
                              hipStream_t stream) {
  const float* x     = (const float*)d_in[0];
  const float* ve    = (const float*)d_in[1];
  const float* cosp  = (const float*)d_in[2];
  const float* sinp  = (const float*)d_in[3];
  const float* Wq    = (const float*)d_in[4];
  const float* Wk    = (const float*)d_in[5];
  const float* Wv    = (const float*)d_in[6];
  const float* Wproj = (const float*)d_in[7];
  const float* Wgate = (const float*)d_in[8];
  float* out = (float*)d_out;

  const int M = BB * TT;            // 4096 rows
  const size_t nX  = (size_t)M * EE;          // 4,194,304
  const size_t nWq = (size_t)HH * DD * EE;    // 1,048,576
  const size_t nWk = (size_t)HKV * DD * EE;   //   524,288
  const size_t nWp = (size_t)EE * EE;         // 1,048,576

  char* w = (char*)d_ws;
  size_t off = 0;
  auto take = [&](size_t bytes) { char* p = w + off; off = (off + bytes + 255) & ~(size_t)255; return p; };
  bf16*  xb   = (bf16*)take(nX * 2);
  bf16*  wqb  = (bf16*)take(nWq * 2);
  bf16*  wkb  = (bf16*)take(nWk * 2);
  bf16*  wvb  = (bf16*)take(nWk * 2);
  bf16*  wpb  = (bf16*)take(nWp * 2);
  float* qraw = (float*)take((size_t)M * HH * DD * 4);
  float* kraw = (float*)take((size_t)M * HKV * DD * 4);
  float* vraw = (float*)take((size_t)M * HKV * DD * 4);
  bf16*  qbn  = (bf16*)take((size_t)M * HH * DD * 2);
  bf16*  kbn  = (bf16*)take((size_t)M * HKV * DD * 2);
  bf16*  vbn  = (bf16*)take((size_t)M * HKV * DD * 2);
  bf16*  ybn  = (bf16*)take((size_t)M * HH * DD * 2);

  // 1) convert activations + weights to bf16
  cvt_bf16_kernel<<<(int)((nX  + 255) / 256), 256, 0, stream>>>(x,     xb,  (int)nX);
  cvt_bf16_kernel<<<(int)((nWq + 255) / 256), 256, 0, stream>>>(Wq,    wqb, (int)nWq);
  cvt_bf16_kernel<<<(int)((nWk + 255) / 256), 256, 0, stream>>>(Wk,    wkb, (int)nWk);
  cvt_bf16_kernel<<<(int)((nWk + 255) / 256), 256, 0, stream>>>(Wv,    wvb, (int)nWk);
  cvt_bf16_kernel<<<(int)((nWp + 255) / 256), 256, 0, stream>>>(Wproj, wpb, (int)nWp);

  // 2) QKV projections (WMMA bf16 GEMMs)
  gemm_bf16_nt<<<dim3((HH * DD) / 128, M / 128), 256, 0, stream>>>(xb, wqb, qraw, M, HH * DD, EE);
  gemm_bf16_nt<<<dim3((HKV * DD) / 128, M / 128), 256, 0, stream>>>(xb, wkb, kraw, M, HKV * DD, EE);
  gemm_bf16_nt<<<dim3((HKV * DD) / 128, M / 128), 256, 0, stream>>>(xb, wvb, vraw, M, HKV * DD, EE);

  // 3) RoPE/RMSNorm on q,k; gate on v
  qpost_kernel<<<(M * HH) / 256, 256, 0, stream>>>(qraw, cosp, sinp, qbn);
  kvpost_kernel<<<(M * HKV) / 256, 256, 0, stream>>>(kraw, vraw, x, ve, Wgate,
                                                     cosp, sinp, kbn, vbn);

  // 4) sliding-window attention (WMMA)
  attn_kernel<<<BB * HH * (TT / 128), 256, 0, stream>>>(qbn, kbn, vbn, ybn);

  // 5) output projection into d_out (fp32)
  gemm_bf16_nt<<<dim3(EE / 128, M / 128), 256, 0, stream>>>(ybn, wpb, out, M, EE, EE);

  (void)in_sizes; (void)n_in; (void)out_size; (void)ws_size;
}
